// GNNWithResiduals_79998060855411
// MI455X (gfx1250) — matrix operand order repaired and checked by reading, not compiled, not run on hardware
//
#include <hip/hip_runtime.h>
#include <hip/hip_bf16.h>

typedef __attribute__((ext_vector_type(16))) _Float16 v16h;
typedef __attribute__((ext_vector_type(8)))  _Float16 v8h;
typedef __attribute__((ext_vector_type(8)))  float    v8f;

#define NGRAPHS 256
#define BN_EPS 1e-5f

// ---------------------------------------------------------------------------
// elementwise / setup kernels
// ---------------------------------------------------------------------------
__global__ void k_fill(float* __restrict__ p, float v, int n) {
    int i = blockIdx.x * blockDim.x + threadIdx.x;
    if (i < n) p[i] = v;
}

__global__ void k_degree(const int* __restrict__ ei, float* __restrict__ deg, int E) {
    int e = blockIdx.x * blockDim.x + threadIdx.x;
    if (e < E) atomicAdd(&deg[ei[E + e]], 1.0f);
}

__global__ void k_rsqrt(const float* __restrict__ deg, float* __restrict__ dinv, int n) {
    int i = blockIdx.x * blockDim.x + threadIdx.x;
    if (i < n) dinv[i] = rsqrtf(deg[i]);
}

__global__ void k_cvt(const float* __restrict__ s, _Float16* __restrict__ d, int n) {
    int i = blockIdx.x * blockDim.x + threadIdx.x;
    if (i < n) d[i] = (_Float16)s[i];
}

// w is [K x N] row-major f32 -> wt is [N x K] row-major f16 (transposed)
__global__ void k_cvt_tr(const float* __restrict__ w, _Float16* __restrict__ wt, int K, int N) {
    int t = blockIdx.x * blockDim.x + threadIdx.x;
    if (t < K * N) {
        int k = t / N, n = t % N;
        wt[(size_t)n * K + k] = (_Float16)w[t];
    }
}

// ---------------------------------------------------------------------------
// Layer 0 (din == 3): fused plain-f32 GEMM for conv (h) and residual paths.
// WMMA is pointless at K=3; this keeps the matrix path branch-free elsewhere.
// ---------------------------------------------------------------------------
__global__ void k_layer0_gemm(const float* __restrict__ x,  const float* __restrict__ wc,
                              const float* __restrict__ wr, const float* __restrict__ rb,
                              float* __restrict__ h, float* __restrict__ res, int n)
{
    int t = blockIdx.x * blockDim.x + threadIdx.x;
    if (t < n * 32) {
        int i = t >> 5, c = t & 31;
        float x0 = x[i * 3 + 0], x1 = x[i * 3 + 1], x2 = x[i * 3 + 2];
        h[t]   = x0 * wc[c] + x1 * wc[32 + c] + x2 * wc[64 + c];
        res[t] = x0 * wr[c] + x1 * wr[32 + c] + x2 * wr[64 + c] + rb[c];
    }
}

// ---------------------------------------------------------------------------
// WMMA tile GEMM: C[M,Npad] = A[M,K](f16,row) * Bt[Npad,K](f16, transposed) + bias
// Compile-time K (full unroll, pure load->wmma clauses) and RELU.
// Preconditions enforced by the host: M % 16 == 0, Npad % 16 == 0, K % 32 == 0,
// Bt/bias zero-padded to Npad. One 16x16 C tile per wave32, 4 waves per block.
// A lane layout (ISA 7.12.2, 16-bit A): lanes 0-15: M=lane, halves 0..7 = K+0..7,
// halves 8..15 = K+16..23; lanes 16-31: M=lane-16, K offsets +8 / +24.
// B (mirror): lanes 0-15: N=lane, halves 0..7 = K+0..7, halves 8..15 = K+16..23; lanes 16-31: N=lane-16, K offsets +8 / +24.
// C: lane<16: col=lane rows 0..7 = acc[0..7]; lane>=16: col=lane-16 rows 8..15.
// ---------------------------------------------------------------------------
template <int K, int RELU>
__global__ void __launch_bounds__(128)
k_gemm_wmma(const _Float16* __restrict__ A, const _Float16* __restrict__ Bt,
            const float* __restrict__ bias, float* __restrict__ C, int M, int Npad)
{
    const int lane = threadIdx.x & 31;
    const int wave = threadIdx.x >> 5;
    const int row0 = (blockIdx.x * 4 + wave) * 16;
    const int col0 = blockIdx.y * 16;
    if (row0 >= M) return;                 // runtime-uniform per wave (tail block only)

    const int hi  = lane >> 4;
    const int l15 = lane & 15;
    const _Float16* __restrict__ aptr = A  + (size_t)(row0 + l15) * K;
    const _Float16* __restrict__ bptr = Bt + (size_t)(col0 + l15) * K;

    v8f acc = {};
#pragma unroll
    for (int k0 = 0; k0 < K; k0 += 32) {
        v8h alo = *(const v8h*)(aptr + k0 + hi * 8);
        v8h aup = *(const v8h*)(aptr + k0 + 16 + hi * 8);
        v8h blo = *(const v8h*)(bptr + k0 + hi * 8);
        v8h bup = *(const v8h*)(bptr + k0 + 16 + hi * 8);
        v16h a = __builtin_shufflevector(alo, aup, 0, 1, 2, 3, 4, 5, 6, 7,
                                         8, 9, 10, 11, 12, 13, 14, 15);
        v16h b = __builtin_shufflevector(blo, bup, 0, 1, 2, 3, 4, 5, 6, 7,
                                         8, 9, 10, 11, 12, 13, 14, 15);
        acc = __builtin_amdgcn_wmma_f32_16x16x32_f16(
                  false, a, false, b, (short)0, acc, false, false);
    }

    const int col = col0 + l15;
    const float bv = bias[col];
    float* __restrict__ cptr = C + (size_t)(row0 + hi * 8) * Npad + col;
#pragma unroll
    for (int r = 0; r < 8; ++r) {
        float v = acc[r] + bv;
        if (RELU) v = fmaxf(v, 0.0f);
        cptr[(size_t)r * Npad] = v;
    }
}

// ---------------------------------------------------------------------------
// edge scatter: accum[dst,:] += h[src,:] * dinv[src]*dinv[dst]
// one thread per (edge, 4 channels); channels contiguous -> coalesced atomics.
// h for the widest layer is 51.2 MB -> L2-resident (192 MB L2).
// ---------------------------------------------------------------------------
__global__ void k_scatter(const float* __restrict__ h, const int* __restrict__ ei,
                          const float* __restrict__ dinv, float* __restrict__ accum,
                          int E, int dout)
{
    const int cpt = dout >> 2;
    const int tid = blockIdx.x * blockDim.x + threadIdx.x;
    if (tid >= E * cpt) return;
    const int e  = tid / cpt;
    const int c0 = (tid % cpt) << 2;
    const int s  = ei[e];
    const int d  = ei[E + e];
    if (c0 == 0) {                          // global_prefetch_b8 a future source row
        int e2 = e + 512;
        if (e2 < E) __builtin_prefetch(h + (size_t)ei[e2] * dout, 0, 1);
    }
    const float nrm = dinv[s] * dinv[d];
    const float4 hv = *(const float4*)(h + (size_t)s * dout + c0);
    float* ap = accum + (size_t)d * dout + c0;
    atomicAdd(ap + 0, hv.x * nrm);
    atomicAdd(ap + 1, hv.y * nrm);
    atomicAdd(ap + 2, hv.z * nrm);
    atomicAdd(ap + 3, hv.w * nrm);
}

// y = accum + h * dinv^2 (self-loop) + conv_b + res   (in-place into accum)
__global__ void k_combine(float* __restrict__ y, const float* __restrict__ h,
                          const float* __restrict__ res, const float* __restrict__ convb,
                          const float* __restrict__ dinv, int n, int dout)
{
    int t = blockIdx.x * blockDim.x + threadIdx.x;
    if (t < n * dout) {
        int i = t / dout, c = t % dout;
        float di = dinv[i];
        y[t] = y[t] + h[t] * di * di + convb[c] + res[t];
    }
}

// per-column sum / sum-of-squares via LDS float atomics (ds_add_f32),
// one global atomic per column per block
__global__ void k_colstats(const float* __restrict__ y, int total, int dout,
                           float* __restrict__ colsum, float* __restrict__ colsq)
{
    __shared__ float s1[256];
    __shared__ float s2[256];
    for (int c = threadIdx.x; c < dout; c += blockDim.x) { s1[c] = 0.0f; s2[c] = 0.0f; }
    __syncthreads();
    for (int t = blockIdx.x * blockDim.x + threadIdx.x; t < total; t += gridDim.x * blockDim.x) {
        float v = y[t];
        int c = t % dout;
        atomicAdd(&s1[c], v);
        atomicAdd(&s2[c], v * v);
    }
    __syncthreads();
    for (int c = threadIdx.x; c < dout; c += blockDim.x) {
        atomicAdd(&colsum[c], s1[c]);
        atomicAdd(&colsq[c],  s2[c]);
    }
}

__global__ void k_bn_relu(const float* __restrict__ y, const float* __restrict__ colsum,
                          const float* __restrict__ colsq, const float* __restrict__ g,
                          const float* __restrict__ b, float* __restrict__ xo,
                          int n, int dout)
{
    int t = blockIdx.x * blockDim.x + threadIdx.x;
    if (t < n * dout) {
        int c = t % dout;
        float inv_n = 1.0f / (float)n;
        float mu  = colsum[c] * inv_n;
        float var = colsq[c] * inv_n - mu * mu;
        float v = g[c] * (y[t] - mu) * rsqrtf(var + BN_EPS) + b[c];
        xo[t] = fmaxf(v, 0.0f);
    }
}

__global__ void k_pool(const float* __restrict__ x, const int* __restrict__ batch,
                       float* __restrict__ psum, float* __restrict__ pcnt, int n)
{
    int t = blockIdx.x * blockDim.x + threadIdx.x;
    if (t < n * 256) {
        int i = t >> 8, c = t & 255;
        int gsel = batch[i];
        atomicAdd(&psum[(size_t)gsel * 256 + c], x[t]);
        if (c == 0) atomicAdd(&pcnt[gsel], 1.0f);
    }
}

__global__ void k_pool_div(const float* __restrict__ psum, const float* __restrict__ pcnt,
                           float* __restrict__ pooled)
{
    int t = blockIdx.x * blockDim.x + threadIdx.x;
    if (t < NGRAPHS * 256) pooled[t] = psum[t] / fmaxf(pcnt[t >> 8], 1.0f);
}

// logits stored with row stride 16 (padded final linear)
__global__ void k_logsoftmax(const float* __restrict__ z, float* __restrict__ out, int G) {
    int g = blockIdx.x * blockDim.x + threadIdx.x;
    if (g < G) {
        float a = z[g * 16 + 0], b = z[g * 16 + 1], c = z[g * 16 + 2];
        float m = fmaxf(a, fmaxf(b, c));
        float l = m + logf(expf(a - m) + expf(b - m) + expf(c - m));
        out[g * 3 + 0] = a - l;
        out[g * 3 + 1] = b - l;
        out[g * 3 + 2] = c - l;
    }
}

// ---------------------------------------------------------------------------
// host driver
// ---------------------------------------------------------------------------
extern "C" void kernel_launch(void* const* d_in, const int* in_sizes, int n_in,
                              void* d_out, int out_size, void* d_ws, size_t ws_size,
                              hipStream_t stream)
{
    const int N = in_sizes[0] / 3;
    const int E = in_sizes[1] / 2;
    const int dims[5] = {3, 32, 64, 128, 256};

    const float* x0    = (const float*)d_in[0];
    const int*   ei    = (const int*)d_in[1];
    const int*   batch = (const int*)d_in[2];

    const float *conv_w[4], *conv_b[4], *res_w[4], *res_b[4], *bn_g[4], *bn_b[4];
    int idx = 3;
    for (int l = 0; l < 4; ++l) {
        conv_w[l] = (const float*)d_in[idx++];
        conv_b[l] = (const float*)d_in[idx++];
        res_w[l]  = (const float*)d_in[idx++];
        res_b[l]  = (const float*)d_in[idx++];
        bn_g[l]   = (const float*)d_in[idx++];
        bn_b[l]   = (const float*)d_in[idx++];
    }
    const float *lin_w[3], *lin_b[3];
    for (int l = 0; l < 3; ++l) {
        lin_w[l] = (const float*)d_in[idx++];
        lin_b[l] = (const float*)d_in[idx++];
    }

    // bump allocator on workspace (256B aligned)
    char* wsp = (char*)d_ws;
    auto alloc = [&](size_t bytes) -> void* {
        void* p = (void*)wsp;
        wsp += (bytes + 255) & ~(size_t)255;
        return p;
    };
    float*     deg    = (float*)alloc((size_t)N * 4);
    float*     dinv   = (float*)alloc((size_t)N * 4);
    float*     hbuf   = (float*)alloc((size_t)N * 256 * 4);
    float*     resb   = (float*)alloc((size_t)N * 256 * 4);
    float*     accum  = (float*)alloc((size_t)N * 256 * 4);
    float*     xcur   = (float*)alloc((size_t)N * 256 * 4);
    _Float16*  xh     = (_Float16*)alloc((size_t)N * 256 * 2);
    _Float16*  wct    = (_Float16*)alloc((size_t)256 * 256 * 2);
    _Float16*  wrt    = (_Float16*)alloc((size_t)256 * 256 * 2);
    float*     zbias  = (float*)alloc(256 * 4);             // zero bias for conv GEMM
    float*     pb2    = (float*)alloc(16 * 4);              // padded lin_b2 (16)
    float*     colsum = (float*)alloc(256 * 4);
    float*     colsq  = (float*)alloc(256 * 4);
    float*     psum   = (float*)alloc((size_t)NGRAPHS * 256 * 4);
    float*     pcnt   = (float*)alloc(NGRAPHS * 4);
    float*     pooled = (float*)alloc((size_t)NGRAPHS * 256 * 4);
    _Float16*  ph     = (_Float16*)alloc((size_t)NGRAPHS * 256 * 2);
    float*     m0     = (float*)alloc((size_t)NGRAPHS * 128 * 4);
    _Float16*  m0h    = (_Float16*)alloc((size_t)NGRAPHS * 128 * 2);
    float*     m1     = (float*)alloc((size_t)NGRAPHS * 64 * 4);
    _Float16*  m1h    = (_Float16*)alloc((size_t)NGRAPHS * 64 * 2);
    float*     logits = (float*)alloc((size_t)NGRAPHS * 16 * 4);

    auto cdiv = [](long long a, long long b) -> int { return (int)((a + b - 1) / b); };

    auto launch_gemm = [&](const _Float16* A, const _Float16* Bt, const float* bias,
                           float* C, int M, int Npad, int K, int relu) {
        dim3 gg(cdiv(M, 64), Npad / 16);
        switch (K * 2 + relu) {
            case  64: k_gemm_wmma< 32, 0><<<gg, 128, 0, stream>>>(A, Bt, bias, C, M, Npad); break;
            case  65: k_gemm_wmma< 32, 1><<<gg, 128, 0, stream>>>(A, Bt, bias, C, M, Npad); break;
            case 128: k_gemm_wmma< 64, 0><<<gg, 128, 0, stream>>>(A, Bt, bias, C, M, Npad); break;
            case 129: k_gemm_wmma< 64, 1><<<gg, 128, 0, stream>>>(A, Bt, bias, C, M, Npad); break;
            case 256: k_gemm_wmma<128, 0><<<gg, 128, 0, stream>>>(A, Bt, bias, C, M, Npad); break;
            case 257: k_gemm_wmma<128, 1><<<gg, 128, 0, stream>>>(A, Bt, bias, C, M, Npad); break;
            case 512: k_gemm_wmma<256, 0><<<gg, 128, 0, stream>>>(A, Bt, bias, C, M, Npad); break;
            case 513: k_gemm_wmma<256, 1><<<gg, 128, 0, stream>>>(A, Bt, bias, C, M, Npad); break;
        }
    };

    // GCN normalization: deg = 1 (self loop) + in-degree; dinv = deg^-1/2
    k_fill<<<cdiv(N, 256), 256, 0, stream>>>(deg, 1.0f, N);
    k_degree<<<cdiv(E, 256), 256, 0, stream>>>(ei, deg, E);
    k_rsqrt<<<cdiv(N, 256), 256, 0, stream>>>(deg, dinv, N);

    hipMemsetAsync(zbias, 0, 256 * 4, stream);

    const float* cur = x0;
    for (int l = 0; l < 4; ++l) {
        const int din = dims[l], dout = dims[l + 1];
        const long long tot = (long long)N * dout;

        if (l == 0) {
            // K=3: plain fused f32 kernel for both conv and residual GEMMs
            k_layer0_gemm<<<cdiv((long long)N * 32, 256), 256, 0, stream>>>(
                x0, conv_w[0], res_w[0], res_b[0], hbuf, resb, N);
        } else {
            const long long nel = (long long)N * din;
            k_cvt<<<cdiv(nel, 256), 256, 0, stream>>>(cur, xh, (int)nel);
            k_cvt_tr<<<cdiv((long long)din * dout, 256), 256, 0, stream>>>(conv_w[l], wct, din, dout);
            k_cvt_tr<<<cdiv((long long)din * dout, 256), 256, 0, stream>>>(res_w[l],  wrt, din, dout);
            launch_gemm(xh, wct, zbias,    hbuf, N, dout, din, 0);
            launch_gemm(xh, wrt, res_b[l], resb, N, dout, din, 0);
        }

        hipMemsetAsync(accum, 0, (size_t)tot * 4, stream);
        const long long sthreads = (long long)E * (dout / 4);
        k_scatter<<<cdiv(sthreads, 256), 256, 0, stream>>>(hbuf, ei, dinv, accum, E, dout);

        k_combine<<<cdiv(tot, 256), 256, 0, stream>>>(accum, hbuf, resb, conv_b[l], dinv, N, dout);

        hipMemsetAsync(colsum, 0, dout * 4, stream);
        hipMemsetAsync(colsq,  0, dout * 4, stream);
        k_colstats<<<512, 256, 0, stream>>>(accum, (int)tot, dout, colsum, colsq);
        k_bn_relu<<<cdiv(tot, 256), 256, 0, stream>>>(accum, colsum, colsq,
                                                      bn_g[l], bn_b[l], xcur, N, dout);
        cur = xcur;
    }

    // global mean pool
    hipMemsetAsync(psum, 0, (size_t)NGRAPHS * 256 * 4, stream);
    hipMemsetAsync(pcnt, 0, NGRAPHS * 4, stream);
    k_pool<<<cdiv((long long)N * 256, 256), 256, 0, stream>>>(xcur, batch, psum, pcnt, N);
    k_pool_div<<<cdiv(NGRAPHS * 256, 256), 256, 0, stream>>>(psum, pcnt, pooled);

    // MLP head (WMMA): 256 -> 128 -> 64 -> 3(pad 16), relu on first two
    k_cvt<<<cdiv(NGRAPHS * 256, 256), 256, 0, stream>>>(pooled, ph, NGRAPHS * 256);
    k_cvt_tr<<<cdiv(256 * 128, 256), 256, 0, stream>>>(lin_w[0], wct, 256, 128);
    launch_gemm(ph, wct, lin_b[0], m0, NGRAPHS, 128, 256, 1);

    k_cvt<<<cdiv(NGRAPHS * 128, 256), 256, 0, stream>>>(m0, m0h, NGRAPHS * 128);
    k_cvt_tr<<<cdiv(128 * 64, 256), 256, 0, stream>>>(lin_w[1], wct, 128, 64);
    launch_gemm(m0h, wct, lin_b[1], m1, NGRAPHS, 64, 128, 1);

    k_cvt<<<cdiv(NGRAPHS * 64, 256), 256, 0, stream>>>(m1, m1h, NGRAPHS * 64);
    hipMemsetAsync(wct, 0, (size_t)16 * 64 * 2, stream);        // zero-padded Bt (16 x 64)
    k_cvt_tr<<<cdiv(64 * 3, 256), 256, 0, stream>>>(lin_w[2], wct, 64, 3);
    hipMemsetAsync(pb2, 0, 16 * 4, stream);                     // zero-padded bias (16)
    hipMemcpyAsync(pb2, lin_b[2], 3 * 4, hipMemcpyDeviceToDevice, stream);
    launch_gemm(m1h, wct, pb2, logits, NGRAPHS, 16, 64, 0);

    k_logsoftmax<<<1, 256, 0, stream>>>(logits, (float*)d_out, NGRAPHS);
}
